// RNN_1597727834310
// MI455X (gfx1250) — matrix operand order repaired and checked by reading, not compiled, not run on hardware
//
#include <hip/hip_runtime.h>
#include <math.h>

// ---------------- CDNA5 WMMA types ----------------
typedef __bf16 bf16_t;
typedef __attribute__((ext_vector_type(16))) __bf16 v16bf;
typedef __attribute__((ext_vector_type(8)))  __bf16 v8bf;
typedef __attribute__((ext_vector_type(8)))  float  v8f;

constexpr int kT  = 512;
constexpr int kB  = 64;
constexpr int kIn = 1024;
constexpr int kH  = 1024;
constexpr int kBH = kB * kH;        // 65536
constexpr int kTB = kT * kB;        // 32768 rows of the hoisted GEMM

// fp32 -> bf16 round-to-nearest-even
static __device__ __forceinline__ bf16_t f2bf(float f) {
  unsigned u = __builtin_bit_cast(unsigned, f);
  unsigned short h = (unsigned short)((u + 0x7FFFu + ((u >> 16) & 1u)) >> 16);
  return __builtin_bit_cast(bf16_t, h);
}

// A-fragment (16x32 bf16, M x K) from row-major bf16 activations.
// ISA 7.12.2: lanes 0-15 (M=lane) hold K 0..7 then 16..23; lanes 16-31 hold K 8..15 then 24..31.
static __device__ __forceinline__ v16bf load_a_bf16(const bf16_t* __restrict__ A, int lda,
                                                    int row0, int k0) {
  const int lane = threadIdx.x & 31;
  const int row  = row0 + (lane & 15);
  const int kb   = k0 + ((lane >> 4) << 3);
  const bf16_t* p = A + (size_t)row * lda + kb;
  v16bf r;
  ((v8bf*)&r)[0] = *(const v8bf*)(p);
  ((v8bf*)&r)[1] = *(const v8bf*)(p + 16);
  return r;
}

// B-fragment (32x16 bf16, K x N) for x @ W^T with W stored [N][K] row-major:
// lane n (0-15) holds W[col0+n][K 0..7, 16..23]; lanes 16-31 hold K 8..15, 24..31. Two 16B loads.
static __device__ __forceinline__ v16bf load_b_bf16(const bf16_t* __restrict__ W, int ldw,
                                                    int col0, int k0) {
  const int lane = threadIdx.x & 31;
  const int col  = col0 + (lane & 15);
  const int kb   = k0 + ((lane >> 4) << 3);
  const bf16_t* p = W + (size_t)col * ldw + kb;
  v16bf r;
  ((v8bf*)&r)[0] = *(const v8bf*)(p);
  ((v8bf*)&r)[1] = *(const v8bf*)(p + 16);
  return r;
}

static __device__ __forceinline__ v8f wmma_bf16(v16bf a, v16bf b, v8f c) {
  return __builtin_amdgcn_wmma_f32_16x16x32_bf16(false, a, false, b, (short)0, c, false, false);
}

// ---------------- elementwise helpers ----------------
__global__ void f2bf_kernel(const float* __restrict__ src, bf16_t* __restrict__ dst, int n) {
  for (int i = blockIdx.x * blockDim.x + threadIdx.x; i < n; i += gridDim.x * blockDim.x)
    dst[i] = f2bf(src[i]);
}

__global__ void bias_add_kernel(const float* __restrict__ a, const float* __restrict__ b,
                                float* __restrict__ dst, int n) {
  for (int i = blockIdx.x * blockDim.x + threadIdx.x; i < n; i += gridDim.x * blockDim.x)
    dst[i] = a[i] + b[i];
}

__global__ void init_h_kernel(const float* __restrict__ h0in, bf16_t* __restrict__ h0bf,
                              bf16_t* __restrict__ h1bf, float* __restrict__ h0f32) {
  for (int i = blockIdx.x * blockDim.x + threadIdx.x; i < kBH; i += gridDim.x * blockDim.x) {
    float v0 = h0in[i];
    float v1 = h0in[kBH + i];
    h0f32[i] = v0;
    h0bf[i]  = f2bf(v0);
    h1bf[i]  = f2bf(v1);
  }
}

__global__ void finalize_kernel(const float* __restrict__ h0f32,
                                const float* __restrict__ out_last,
                                float* __restrict__ hn) {
  for (int i = blockIdx.x * blockDim.x + threadIdx.x; i < kBH; i += gridDim.x * blockDim.x) {
    hn[i]       = h0f32[i];
    hn[kBH + i] = out_last[i];
  }
}

// ---------------- hoisted GEMM: pre0 = Xb @ W_ih0^T + (b_ih0 + b_hh0) ----------------
// Xb: [T*B, IN] bf16. Wave tile: 16 rows x 64 cols. Fully unrolled K with a depth-1
// software pipeline: parity-indexed fragment sets become distinct SSA values (no copies),
// and loads for step i+1 are issued before the WMMAs of step i (overlap, no 0x0 stalls).
__global__ __launch_bounds__(128) void gemm_pre0_kernel(const bf16_t* __restrict__ Xb,
                                                        const bf16_t* __restrict__ Wih0,
                                                        const float* __restrict__ bias0,
                                                        float* __restrict__ pre0) {
  const int lane = threadIdx.x & 31;
  const int wg   = blockIdx.x * 4 + (threadIdx.x >> 5);
  const int m0   = (wg >> 4) * 16;    // 2048 M-tiles
  const int n0   = (wg & 15) * 64;    // 16 N-segments of 64
  v8f acc[4];
#pragma unroll
  for (int a = 0; a < 4; ++a) {
    float bv = bias0[n0 + a * 16 + (lane & 15)];
#pragma unroll
    for (int r = 0; r < 8; ++r) acc[a][r] = bv;
  }

  constexpr int NI = kIn / 32;        // 32 K-steps... (kIn=1024 -> 32 iters of K=32)
  v16bf af[2];
  v16bf bm[2][4];
  af[0] = load_a_bf16(Xb, kIn, m0, 0);
#pragma unroll
  for (int a = 0; a < 4; ++a) bm[0][a] = load_b_bf16(Wih0, kIn, n0 + a * 16, 0);
#pragma unroll
  for (int i = 0; i < NI; ++i) {
    const int cur = i & 1, nxt = cur ^ 1;
    const int kn  = (i + 1 < NI) ? (i + 1) * 32 : 0;   // dummy reload on final step
    af[nxt] = load_a_bf16(Xb, kIn, m0, kn);
#pragma unroll
    for (int a = 0; a < 4; ++a) bm[nxt][a] = load_b_bf16(Wih0, kIn, n0 + a * 16, kn);
#pragma unroll
    for (int a = 0; a < 4; ++a) acc[a] = wmma_bf16(af[cur], bm[cur][a], acc[a]);
  }

  const int rbase = m0 + ((lane >> 4) << 3);
  const int cbase = n0 + (lane & 15);
#pragma unroll
  for (int a = 0; a < 4; ++a)
#pragma unroll
    for (int r = 0; r < 8; ++r)
      pre0[(size_t)(rbase + r) * kH + cbase + a * 16] = acc[a][r];
}

// ---------------- recurrent step, layer 0: h0 = tanh(pre0[t] + h0 @ W_hh0^T) ----------------
// Wave tile: 16 rows x 32 cols -> 128 waves -> 32 blocks. Same pipelined full unroll.
__global__ __launch_bounds__(128) void step_l0_kernel(const bf16_t* __restrict__ h0src,
                                                      const bf16_t* __restrict__ Whh0,
                                                      const float* __restrict__ pre0t,
                                                      bf16_t* __restrict__ h0dst,
                                                      float* __restrict__ h0f32) {
  const int lane  = threadIdx.x & 31;
  const int wg    = blockIdx.x * 4 + (threadIdx.x >> 5);
  const int m0    = (wg >> 5) * 16;
  const int n0    = (wg & 31) * 32;
  const int rbase = m0 + ((lane >> 4) << 3);
  const int cbase = n0 + (lane & 15);
  v8f acc[2];
#pragma unroll
  for (int a = 0; a < 2; ++a)
#pragma unroll
    for (int r = 0; r < 8; ++r)
      acc[a][r] = pre0t[(size_t)(rbase + r) * kH + cbase + a * 16];

  constexpr int NI = kH / 32;
  v16bf af[2];
  v16bf bm[2][2];
  af[0] = load_a_bf16(h0src, kH, m0, 0);
#pragma unroll
  for (int a = 0; a < 2; ++a) bm[0][a] = load_b_bf16(Whh0, kH, n0 + a * 16, 0);
#pragma unroll
  for (int i = 0; i < NI; ++i) {
    const int cur = i & 1, nxt = cur ^ 1;
    const int kn  = (i + 1 < NI) ? (i + 1) * 32 : 0;
    af[nxt] = load_a_bf16(h0src, kH, m0, kn);
#pragma unroll
    for (int a = 0; a < 2; ++a) bm[nxt][a] = load_b_bf16(Whh0, kH, n0 + a * 16, kn);
#pragma unroll
    for (int a = 0; a < 2; ++a) acc[a] = wmma_bf16(af[cur], bm[cur][a], acc[a]);
  }

#pragma unroll
  for (int a = 0; a < 2; ++a)
#pragma unroll
    for (int r = 0; r < 8; ++r) {
      float v = tanhf(acc[a][r]);
      int idx = (rbase + r) * kH + cbase + a * 16;
      h0f32[idx] = v;
      h0dst[idx] = f2bf(v);
    }
}

// ---------------- recurrent step, layer 1: h1 = tanh(h0 @ W_ih1^T + h1 @ W_hh1^T + bias1) ----
__global__ __launch_bounds__(128) void step_l1_kernel(const bf16_t* __restrict__ h0cur,
                                                      const bf16_t* __restrict__ h1src,
                                                      const bf16_t* __restrict__ Wih1,
                                                      const bf16_t* __restrict__ Whh1,
                                                      const float* __restrict__ bias1,
                                                      bf16_t* __restrict__ h1dst,
                                                      float* __restrict__ outT) {
  const int lane  = threadIdx.x & 31;
  const int wg    = blockIdx.x * 4 + (threadIdx.x >> 5);
  const int m0    = (wg >> 5) * 16;
  const int n0    = (wg & 31) * 32;
  const int rbase = m0 + ((lane >> 4) << 3);
  const int cbase = n0 + (lane & 15);
  v8f acc[2];
#pragma unroll
  for (int a = 0; a < 2; ++a) {
    float bv = bias1[cbase + a * 16];
#pragma unroll
    for (int r = 0; r < 8; ++r) acc[a][r] = bv;
  }

  constexpr int NI = kH / 32;
  // ---- GEMM 1: h0cur @ W_ih1^T ----
  {
    v16bf af[2];
    v16bf bm[2][2];
    af[0] = load_a_bf16(h0cur, kH, m0, 0);
#pragma unroll
    for (int a = 0; a < 2; ++a) bm[0][a] = load_b_bf16(Wih1, kH, n0 + a * 16, 0);
#pragma unroll
    for (int i = 0; i < NI; ++i) {
      const int cur = i & 1, nxt = cur ^ 1;
      const int kn  = (i + 1 < NI) ? (i + 1) * 32 : 0;
      af[nxt] = load_a_bf16(h0cur, kH, m0, kn);
#pragma unroll
      for (int a = 0; a < 2; ++a) bm[nxt][a] = load_b_bf16(Wih1, kH, n0 + a * 16, kn);
#pragma unroll
      for (int a = 0; a < 2; ++a) acc[a] = wmma_bf16(af[cur], bm[cur][a], acc[a]);
    }
  }
  // ---- GEMM 2: h1src @ W_hh1^T ----
  {
    v16bf af[2];
    v16bf bm[2][2];
    af[0] = load_a_bf16(h1src, kH, m0, 0);
#pragma unroll
    for (int a = 0; a < 2; ++a) bm[0][a] = load_b_bf16(Whh1, kH, n0 + a * 16, 0);
#pragma unroll
    for (int i = 0; i < NI; ++i) {
      const int cur = i & 1, nxt = cur ^ 1;
      const int kn  = (i + 1 < NI) ? (i + 1) * 32 : 0;
      af[nxt] = load_a_bf16(h1src, kH, m0, kn);
#pragma unroll
      for (int a = 0; a < 2; ++a) bm[nxt][a] = load_b_bf16(Whh1, kH, n0 + a * 16, kn);
#pragma unroll
      for (int a = 0; a < 2; ++a) acc[a] = wmma_bf16(af[cur], bm[cur][a], acc[a]);
    }
  }

#pragma unroll
  for (int a = 0; a < 2; ++a)
#pragma unroll
    for (int r = 0; r < 8; ++r) {
      float v = tanhf(acc[a][r]);
      int idx = (rbase + r) * kH + cbase + a * 16;
      outT[idx]  = v;
      h1dst[idx] = f2bf(v);
    }
}

// ---------------- host-side launch ----------------
extern "C" void kernel_launch(void* const* d_in, const int* in_sizes, int n_in,
                              void* d_out, int out_size, void* d_ws, size_t ws_size,
                              hipStream_t stream) {
  (void)in_sizes; (void)n_in; (void)out_size; (void)ws_size;
  const float* input = (const float*)d_in[0];   // [T,B,IN]
  const float* h0in  = (const float*)d_in[1];   // [2,B,H]
  const float* W_ih0 = (const float*)d_in[2];
  const float* b_ih0 = (const float*)d_in[3];
  const float* W_hh0 = (const float*)d_in[4];
  const float* b_hh0 = (const float*)d_in[5];
  const float* W_ih1 = (const float*)d_in[6];
  const float* b_ih1 = (const float*)d_in[7];
  const float* W_hh1 = (const float*)d_in[8];
  const float* b_hh1 = (const float*)d_in[9];
  float* out = (float*)d_out;                   // [T*B*H] outputs + [2*B*H] h_n

  // workspace carve-up
  char* ws = (char*)d_ws;
  size_t off = 0;
  auto carve = [&](size_t bytes) -> void* {
    void* p = ws + off;
    off = (off + bytes + 255) & ~(size_t)255;
    return p;
  };
  bf16_t* Wih0b = (bf16_t*)carve((size_t)kH * kIn * sizeof(bf16_t));
  bf16_t* Whh0b = (bf16_t*)carve((size_t)kH * kH  * sizeof(bf16_t));
  bf16_t* Wih1b = (bf16_t*)carve((size_t)kH * kH  * sizeof(bf16_t));
  bf16_t* Whh1b = (bf16_t*)carve((size_t)kH * kH  * sizeof(bf16_t));
  float*  bias0 = (float*)carve(kH * sizeof(float));
  float*  bias1 = (float*)carve(kH * sizeof(float));
  float*  h0f32 = (float*)carve((size_t)kBH * sizeof(float));
  bf16_t* h0A   = (bf16_t*)carve((size_t)kBH * sizeof(bf16_t));
  bf16_t* h0B   = (bf16_t*)carve((size_t)kBH * sizeof(bf16_t));
  bf16_t* h1A   = (bf16_t*)carve((size_t)kBH * sizeof(bf16_t));
  bf16_t* h1B   = (bf16_t*)carve((size_t)kBH * sizeof(bf16_t));
  bf16_t* Xb    = (bf16_t*)carve((size_t)kTB * kIn * sizeof(bf16_t));   // 64 MB
  float*  pre0  = (float*)carve((size_t)kTB * kH * sizeof(float));      // 128 MB

  // 1) weight / bias / input prep (parallel, cheap)
  f2bf_kernel<<<1024, 256, 0, stream>>>(W_ih0, Wih0b, kH * kIn);
  f2bf_kernel<<<1024, 256, 0, stream>>>(W_hh0, Whh0b, kH * kH);
  f2bf_kernel<<<1024, 256, 0, stream>>>(W_ih1, Wih1b, kH * kH);
  f2bf_kernel<<<1024, 256, 0, stream>>>(W_hh1, Whh1b, kH * kH);
  f2bf_kernel<<<4096, 256, 0, stream>>>(input, Xb, kTB * kIn);
  bias_add_kernel<<<4, 256, 0, stream>>>(b_ih0, b_hh0, bias0, kH);
  bias_add_kernel<<<4, 256, 0, stream>>>(b_ih1, b_hh1, bias1, kH);
  init_h_kernel<<<128, 256, 0, stream>>>(h0in, h0A, h1A, h0f32);

  // 2) hoisted input GEMM over all timesteps: pre0 = Xb @ W_ih0^T + bias0
  gemm_pre0_kernel<<<(kTB / 16) * 16 / 4, 128, 0, stream>>>(Xb, Wih0b, bias0, pre0);

  // 3) sequential recurrence (bf16 h ping-pong)
  for (int t = 0; t < kT; ++t) {
    const bf16_t* h0s = (t & 1) ? h0B : h0A;
    bf16_t*       h0d = (t & 1) ? h0A : h0B;
    const bf16_t* h1s = (t & 1) ? h1B : h1A;
    bf16_t*       h1d = (t & 1) ? h1A : h1B;
    step_l0_kernel<<<32, 128, 0, stream>>>(h0s, Whh0b, pre0 + (size_t)t * kBH, h0d, h0f32);
    step_l1_kernel<<<32, 128, 0, stream>>>(h0d, h1s, Wih1b, Whh1b, bias1, h1d,
                                           out + (size_t)t * kBH);
  }

  // 4) h_n = [h0_final, h1_final]
  finalize_kernel<<<128, 256, 0, stream>>>(h0f32, out + (size_t)(kT - 1) * kBH,
                                           out + (size_t)kT * kBH);
}
